// LLICTIEntropyModel4_89721866813457
// MI455X (gfx1250) — matrix operand order repaired and checked by reading, not compiled, not run on hardware
//
#include <hip/hip_runtime.h>
#include <hip/hip_bf16.h>
#include <math.h>

// Fused LLIC entropy-model head for MI455X (gfx1250, wave32, WMMA).
// One workgroup (8 wave32) = 8x16 pixel tile. All activations stay in LDS;
// convs are v_wmma_f32_16x16x32_f16 GEMMs fed by vectorized ds_load_b128;
// only the final int16 CDF table (33.7 MB) is streamed to HBM.

typedef __attribute__((ext_vector_type(16))) _Float16 v16h;
typedef __attribute__((ext_vector_type(8)))  _Float16 v8h;
typedef __attribute__((ext_vector_type(8)))  float    v8f;

#define HTILE 8
#define WTILE 16
#define NPIX  (HTILE * WTILE)   // 128 pixels per workgroup
#define C0    192
#define CG    64
#define NGRP  3
#define INH   (HTILE + 5)       // 13 padded rows
#define INW   (WTILE + 5)       // 21 padded cols
#define INS   22                // s_in row stride
#define LP    257
#define PATS  72                // im2col row stride (halves): 36 dw, gcd(36,64)=4
#define H0S   200               // h0 row stride (halves): 100 dw, gcd(100,64)=4
#define H1S   72                // h1 row stride (halves)

// ---- CDNA5 wave32 WMMA fragment loaders (contiguous k order, same for A and B) ----
// B (32x16 f16): lane holds N=lane&15; element i -> K = (lane&16?16:0)+i
// => 16 contiguous halves starting at 32*s + (lane&16?16:0).
__device__ __forceinline__ v16h ldB(const _Float16* base16aligned) {
  const v8h lo = *(const v8h*)(base16aligned);
  const v8h hi = *(const v8h*)(base16aligned + 8);
  v16h r;
#pragma unroll
  for (int i = 0; i < 8; ++i) { r[i] = lo[i]; r[i + 8] = hi[i]; }
  return r;
}
// A (16x32 f16): lane holds M=lane&15; element i -> K = (lane&16?16:0) + i
// => 16 contiguous halves at k0, k0 = 32*s + (lane&16?16:0).
__device__ __forceinline__ v16h ldA(const _Float16* row, int s, int lane) {
  const int k0 = 32 * s + ((lane & 16) ? 16 : 0);
  const v8h lo = *(const v8h*)(row + k0);
  const v8h hi = *(const v8h*)(row + k0 + 8);
  v16h r;
#pragma unroll
  for (int i = 0; i < 8; ++i) { r[i] = lo[i]; r[i + 8] = hi[i]; }
  return r;
}

__device__ __forceinline__ float ndtr_f(float z) {
  return 0.5f * erfcf(-z * 0.70710678118654752f);
}

__global__ void __launch_bounds__(256)
entropy_fused_kernel(const float* __restrict__ yc,
                     const float* __restrict__ W0, const float* __restrict__ b0,
                     const float* __restrict__ W1, const float* __restrict__ b1,
                     const float* __restrict__ W2, const float* __restrict__ b2,
                     short* __restrict__ out)
{
  __shared__ float    s_in[INH * INS];          // padded input tile (f32)
  __shared__ _Float16 s_w0[C0 * 64];            // W0, K padded 36->64
  __shared__ _Float16 s_w1[C0 * 64];            // W1 grouped 1x1 (192 x 64)
  __shared__ _Float16 s_w2[NGRP * 16 * 64];     // W2 per group, M padded 3->16
  __shared__ float    s_b0[C0];
  __shared__ float    s_b1[C0];
  __shared__ float    s_b2[16];
  __shared__ _Float16 s_pat[NPIX * PATS];       // im2col patches [p][k]  18 KB
  __shared__ _Float16 s_h0[NPIX * H0S];         // relu(conv0)   [p][c]  50 KB
  __shared__ _Float16 s_h1[NPIX * H1S];         // relu(conv1)   [p][c]  18 KB
  __shared__ float    s_p[9 * NPIX];            // mu(0..2) sr(3..5) wr(6..8)

  const int tid  = threadIdx.x;
  const int lane = tid & 31;
  const int wv   = tid >> 5;                    // 8 wave32
  const int hi8  = (lane & 16) ? 8 : 0;
  const int kb16 = (lane & 16) ? 16 : 0;
  const int gx0  = blockIdx.x * WTILE;
  const int gy0  = blockIdx.y * HTILE;

  __builtin_prefetch(yc + (size_t)gy0 * 256 + gx0, 0, 0);   // global_prefetch_b8

  // ---- Stage weights to LDS (f32 -> f16, zero-pad K / M) ----
  for (int i = tid; i < C0 * 64; i += 256) {
    const int c = i >> 6, k = i & 63;
    s_w0[i] = (_Float16)(k < 36 ? W0[c * 36 + k] : 0.f);
    s_w1[i] = (_Float16)W1[i];
  }
  for (int i = tid; i < NGRP * 16 * 64; i += 256) {
    const int g = i >> 10, r = i & 1023, m = r >> 6, k = r & 63;
    s_w2[i] = (_Float16)(m < 3 ? W2[(g * 3 + m) * 64 + k] : 0.f);
  }
  if (tid < C0) { s_b0[tid] = b0[tid]; s_b1[tid] = b1[tid]; }
  if (tid < 16) s_b2[tid] = (tid < 9) ? b2[tid] : 0.f;

  // ---- Stage edge-clamped padded input tile (13 x 21) ----
  for (int i = tid; i < INH * INW; i += 256) {
    const int r = i / INW, c = i % INW;
    const int y = min(max(gy0 + r - 2, 0), 255);
    const int x = min(max(gx0 + c - 2, 0), 255);
    s_in[r * INS + c] = yc[y * 256 + x];
  }
  __syncthreads();

  // ---- Build im2col patch matrix once: pat[p][k], K padded to 64 ----
  for (int i = tid; i < NPIX * 64; i += 256) {
    const int p = i >> 6, k = i & 63;
    const int py = p >> 4, px = p & 15;
    const float v = (k < 36) ? s_in[(py + k / 6) * INS + (px + k % 6)] : 0.f;
    s_pat[p * PATS + k] = (_Float16)v;
  }
  __syncthreads();

  // ================ conv0: h0[p][192] = W0[192 x K64] * pat ================
  for (int t = wv; t < 12 * (NPIX / 16); t += 8) {       // 96 output tiles
    const int mt = t >> 3, nt = t & 7;
    const int m  = mt * 16 + (lane & 15);
    const int p  = nt * 16 + (lane & 15);
    const _Float16* arow = s_w0 + m * 64;
    const _Float16* brow = s_pat + p * PATS;
    v8f acc = {};
#pragma unroll
    for (int s = 0; s < 2; ++s) {
      const v16h a = ldA(arow, s, lane);
      const v16h b = ldB(brow + 32 * s + kb16);
      acc = __builtin_amdgcn_wmma_f32_16x16x32_f16(false, a, false, b,
                                                   (short)0, acc, false, false);
    }
    const int chb = mt * 16 + hi8;                       // consecutive channels
    v8h hv;
#pragma unroll
    for (int r = 0; r < 8; ++r)
      hv[r] = (_Float16)fmaxf(acc[r] + s_b0[chb + r], 0.f);
    *(v8h*)&s_h0[p * H0S + chb] = hv;                    // one ds_store_b128
  }
  __syncthreads();

  // ============ grouped conv1 (relu) + conv2, per group ====================
  for (int g = 0; g < NGRP; ++g) {
    // conv1: h1[p][64] = W1g[64 x 64] * h0g
    for (int t = wv; t < 4 * 8; t += 8) {
      const int mt = t >> 3, nt = t & 7;
      const int p  = nt * 16 + (lane & 15);
      const _Float16* arow = s_w1 + (g * CG + mt * 16 + (lane & 15)) * 64;
      const _Float16* brow = s_h0 + p * H0S + g * CG;
      v8f acc = {};
#pragma unroll
      for (int s = 0; s < 2; ++s) {
        const v16h a = ldA(arow, s, lane);
        const v16h b = ldB(brow + 32 * s + kb16);
        acc = __builtin_amdgcn_wmma_f32_16x16x32_f16(false, a, false, b,
                                                     (short)0, acc, false, false);
      }
      const int cmb = mt * 16 + hi8;
      v8h hv;
#pragma unroll
      for (int r = 0; r < 8; ++r)
        hv[r] = (_Float16)fmaxf(acc[r] + s_b1[g * CG + cmb + r], 0.f);
      *(v8h*)&s_h1[p * H1S + cmb] = hv;
    }
    __syncthreads();

    // conv2: p[3 x 128] (M padded to 16) = W2g[16 x 64] * h1
    {
      const int nt = wv;                                 // 8 waves == 8 N-tiles
      const int p  = nt * 16 + (lane & 15);
      const _Float16* arow = s_w2 + g * 1024 + (lane & 15) * 64;
      const _Float16* brow = s_h1 + p * H1S;
      v8f acc = {};
#pragma unroll
      for (int s = 0; s < 2; ++s) {
        const v16h a = ldA(arow, s, lane);
        const v16h b = ldB(brow + 32 * s + kb16);
        acc = __builtin_amdgcn_wmma_f32_16x16x32_f16(false, a, false, b,
                                                     (short)0, acc, false, false);
      }
#pragma unroll
      for (int r = 0; r < 8; ++r) {
        const int mr = r + hi8;
        if (mr < 3)
          s_p[(g * 3 + mr) * NPIX + p] = acc[r] + s_b2[g * 3 + mr];
      }
    }
    __syncthreads();
  }

  // ================= GMM CDF tail: 257 points per pixel ====================
  // Wave wv owns pixels [wv*16, wv*16+16); lanes sweep l = 32j+lane so every
  // global_store_b16 is a contiguous 64-byte burst (output phase is BW-bound).
  for (int pp = 0; pp < 16; ++pp) {
    const int p = wv * 16 + pp;
    const float mu0 = s_p[0 * NPIX + p], mu1 = s_p[1 * NPIX + p], mu2 = s_p[2 * NPIX + p];
    const float sr0 = s_p[3 * NPIX + p], sr1 = s_p[4 * NPIX + p], sr2 = s_p[5 * NPIX + p];
    const float wr0 = s_p[6 * NPIX + p], wr1 = s_p[7 * NPIX + p], wr2 = s_p[8 * NPIX + p];

    const float is0 = 1.f / fmaxf(expf(sr0), 0.11f);
    const float is1 = 1.f / fmaxf(expf(sr1), 0.11f);
    const float is2 = 1.f / fmaxf(expf(sr2), 0.11f);

    const float wm = fmaxf(wr0, fmaxf(wr1, wr2));
    const float e0 = expf(wr0 - wm), e1 = expf(wr1 - wm), e2 = expf(wr2 - wm);
    const float inv = 1.f / (e0 + e1 + e2);
    const float wt0 = e0 * inv, wt1 = e1 * inv, wt2 = e2 * inv;

    const int py = p >> 4, px = p & 15;
    short* op = out + (((size_t)(gy0 + py) * 256 + (gx0 + px)) * LP);

#pragma unroll
    for (int j = 0; j < 9; ++j) {
      const int l = j * 32 + lane;
      if (l < LP) {
        float t = (float)l * (1.f / 255.f) - (127.5f / 255.f);
        if (l == 0)   t = -147.5f / 255.f;
        if (l == 256) t =  148.5f / 255.f;
        float c = wt0 * ndtr_f((t - mu0) * is0)
                + wt1 * ndtr_f((t - mu1) * is1)
                + wt2 * ndtr_f((t - mu2) * is2);
        c = fminf(fmaxf(c, 0.f), 1.f - 1e-6f);
        const int q = (int)rintf(c * 65280.f);   // 2^16 - (LP-1)
        op[l] = (short)(q + l);                  // int16 wrap matches numpy
      }
    }
  }
}

extern "C" void kernel_launch(void* const* d_in, const int* in_sizes, int n_in,
                              void* d_out, int out_size, void* d_ws, size_t ws_size,
                              hipStream_t stream) {
  (void)in_sizes; (void)n_in; (void)d_ws; (void)ws_size; (void)out_size;
  const float* yc = (const float*)d_in[0];
  const float* W0 = (const float*)d_in[1];
  const float* b0 = (const float*)d_in[2];
  const float* W1 = (const float*)d_in[3];
  const float* b1 = (const float*)d_in[4];
  const float* W2 = (const float*)d_in[5];
  const float* b2 = (const float*)d_in[6];
  short* out = (short*)d_out;                    // reference output dtype: int16

  dim3 grid(256 / WTILE, 256 / HTILE);           // 16 x 32 = 512 workgroups
  dim3 block(256);                               // 8 wave32
  entropy_fused_kernel<<<grid, block, 0, stream>>>(yc, W0, b0, W1, b1, W2, b2, out);
}
